// FlaxMistralAttention_52209622450424
// MI455X (gfx1250) — matrix operand order repaired and checked by reading, not compiled, not run on hardware
//
#include <hip/hip_runtime.h>
#include <hip/hip_bf16.h>
#include <stdint.h>

// ---------------------------------------------------------------------------
// Mistral attention block for MI455X (gfx1250, wave32, WMMA).
//   hidden(1,2048,4096) f32; Wq(4096,4096) Wk/Wv(4096,1024) Wo(4096,4096) f32
//   out(1,2048,4096) f32
// bf16 operands + f32 WMMA accumulation; flash-style attention (scores never
// hit HBM); causal mask applied analytically (== the additive -1e9 mask).
// GEMM A-tiles staged to LDS with gfx1250 async global->LDS copies
// (ASYNCcnt), W-tiles staged transposed for conflict-free ds_load_b128.
// ---------------------------------------------------------------------------

typedef __attribute__((ext_vector_type(16))) __bf16 bf16x16;
typedef __attribute__((ext_vector_type(8)))  __bf16 bf16x8;
typedef __attribute__((ext_vector_type(4)))  __bf16 bf16x4;
typedef __attribute__((ext_vector_type(8)))  float  f32x8;

#define S_LEN   2048
#define H_DIM   4096
#define NH      32
#define NKV     8
#define HD      128
#define NEG_INF (-1e9f)

static __device__ __forceinline__ bf16x16 concat8(bf16x8 lo, bf16x8 hi) {
  return __builtin_shufflevector(lo, hi, 0,1,2,3,4,5,6,7,8,9,10,11,12,13,14,15);
}
static __device__ __forceinline__ f32x8 zero8() {
  f32x8 z;
#pragma unroll
  for (int i = 0; i < 8; ++i) z[i] = 0.0f;
  return z;
}

// gfx1250 async global->LDS copy: each lane moves 16B. VDST = per-lane LDS
// byte address (low 32 bits of the flat shared pointer), VADDR = 64-bit
// global address. Tracked by ASYNCcnt.
static __device__ __forceinline__ void async_copy_b128(const __bf16* gsrc, __bf16* lds_dst) {
  uint32_t dst = (uint32_t)(uintptr_t)lds_dst;
  uint64_t src = (uint64_t)(uintptr_t)gsrc;
  asm volatile("global_load_async_to_lds_b128 %0, %1, off" : : "v"(dst), "v"(src) : "memory");
}
static __device__ __forceinline__ void wait_asynccnt0() {
  asm volatile("s_wait_asynccnt 0x0" ::: "memory");
}

// ---------------------------------------------------------------------------
// fp32 -> bf16 conversion, 4 elements/thread (all sizes are multiples of 4)
// ---------------------------------------------------------------------------
__global__ void cvt_f32_bf16(const float* __restrict__ in, __bf16* __restrict__ out, int n4) {
  int i = blockIdx.x * 256 + threadIdx.x;
  if (i >= n4) return;
  float4 v = ((const float4*)in)[i];
  bf16x4 o;
  o[0] = (__bf16)v.x; o[1] = (__bf16)v.y; o[2] = (__bf16)v.z; o[3] = (__bf16)v.w;
  ((bf16x4*)out)[i] = o;
}

// ---------------------------------------------------------------------------
// C(MxN) = A(MxK,bf16) @ W(KxN,bf16), f32 accumulation.
// Block = 256 threads (8 waves); per k-chunk of 32:
//  * A block-tile (128 rows x 32 k) staged via async global->LDS b128 into
//    80B-padded rows (16B-aligned, bank-conflict-free ds_load_b128 reads).
//  * W chunk (32k x 64n) staged transposed in LDS: Ws[n][k], 80B rows.
// Wave w owns rows [16w,16w+16), 4 n-tiles sharing one A fragment.
// ---------------------------------------------------------------------------
template <bool WRITE_F32>
__global__ __launch_bounds__(256) void gemm_bf16_kernel(
    const __bf16* __restrict__ A, const __bf16* __restrict__ W,
    float* __restrict__ Cf, __bf16* __restrict__ Cb, int M, int N, int K) {
  __shared__ alignas(16) __bf16 Ws[64][40];   // W chunk, transposed
  __shared__ alignas(16) __bf16 As[128][40];  // A tile, padded rows

  const int lane = threadIdx.x & 31;
  const int wv   = threadIdx.x >> 5;
  const int hf   = lane >> 4;   // 0/1: lane half
  const int l16  = lane & 15;
  const int bm0  = blockIdx.x * 128;
  const int n0   = blockIdx.y * 64;

  f32x8 acc[4];
#pragma unroll
  for (int j = 0; j < 4; ++j) acc[j] = zero8();

  for (int k0 = 0; k0 < K; k0 += 32) {
    // async stage of A[bm0..bm0+128)[k0..k0+32): 512 x 16B granules
    {
#pragma unroll
      for (int g = 0; g < 2; ++g) {
        int id  = threadIdx.x + 256 * g;  // 0..511
        int row = id >> 2;                // 0..127
        int seg = id & 3;                 // 16B segment within 64B row
        async_copy_b128(A + (size_t)(bm0 + row) * K + k0 + seg * 8, &As[row][seg * 8]);
      }
    }
    // cooperative transposed stage of W[k0..k0+32)[n0..n0+64)
    {
      int kk = threadIdx.x >> 3;        // 0..31
      int nn = (threadIdx.x & 7) * 8;   // 0..56
      bf16x8 w = *(const bf16x8*)(W + (size_t)(k0 + kk) * N + n0 + nn);
#pragma unroll
      for (int i = 0; i < 8; ++i) Ws[nn + i][kk] = w[i];
    }
    wait_asynccnt0();   // own async LDS writes complete
    __syncthreads();    // all waves' stages visible

    // A fragment from LDS: elem e -> k = (e/8)*16 + hf*8 + e%8
    const __bf16* ap = &As[wv * 16 + l16][hf * 8];
    bf16x16 afrag = concat8(*(const bf16x8*)ap, *(const bf16x8*)(ap + 16));

#pragma unroll
    for (int j = 0; j < 4; ++j) {
      // B fragment: lane -> col n0+16j+l16; elem e -> k = (e/8)*16 + hf*8 + e%8
      const __bf16* bp = &Ws[j * 16 + l16][hf * 8];
      bf16x16 bfrag = concat8(*(const bf16x8*)bp, *(const bf16x8*)(bp + 16));
      acc[j] = __builtin_amdgcn_wmma_f32_16x16x32_bf16(
          false, afrag, false, bfrag, (short)0, acc[j], false, false);
    }
    __syncthreads();
  }

  // C/D layout: VGPR r -> row m = hf*8 + r, col n = l16
#pragma unroll
  for (int j = 0; j < 4; ++j) {
    int n = n0 + j * 16 + l16;
#pragma unroll
    for (int r = 0; r < 8; ++r) {
      size_t idx = (size_t)(bm0 + wv * 16 + hf * 8 + r) * N + n;
      if (WRITE_F32) Cf[idx] = acc[j][r];
      else           Cb[idx] = (__bf16)acc[j][r];
    }
  }
}

// ---------------------------------------------------------------------------
// In-place RoPE on bf16 (S, nheads*128). One thread per (s, h, j<64) pair:
// reads x[j], x[j+64] then writes both -> no cross-thread hazard.
// ---------------------------------------------------------------------------
__global__ void rope_kernel(__bf16* __restrict__ X, const int* __restrict__ pos_ids, int nheads) {
  int idx = blockIdx.x * 256 + threadIdx.x;
  int total = S_LEN * nheads * 64;
  if (idx >= total) return;
  int j = idx & 63;
  int h = (idx >> 6) % nheads;
  int s = idx / (64 * nheads);
  float pos = (float)pos_ids[s];
  float inv = powf(10000.0f, -(float)(2 * j) / 128.0f);
  float ang = pos * inv;
  float c = cosf(ang), sn = sinf(ang);
  size_t base = (size_t)s * nheads * HD + h * HD + j;
  float x1 = (float)X[base];
  float x2 = (float)X[base + 64];
  X[base]      = (__bf16)(x1 * c - x2 * sn);
  X[base + 64] = (__bf16)(x2 * c + x1 * sn);
}

// V (S, NKV*HD) -> Vt (NKV, HD, S): makes P@V B-fragments contiguous loads.
__global__ void v_transpose_kernel(const __bf16* __restrict__ V, __bf16* __restrict__ Vt) {
  int idx = blockIdx.x * 256 + threadIdx.x;
  if (idx >= S_LEN * NKV * HD) return;
  int c = idx & 1023;
  int s = idx >> 10;
  int kvh = c >> 7, d = c & 127;
  Vt[((size_t)kvh * HD + d) * S_LEN + s] = V[idx];
}

// ---------------------------------------------------------------------------
// Flash attention: one wave per (head, 16-query tile). Online softmax in f32;
// P converted C-layout -> A-layout through a padded per-wave LDS tile
// (wave-private LDS ops are in-order on CDNA5).
// ---------------------------------------------------------------------------
__global__ __launch_bounds__(256) void flash_attn_kernel(
    const __bf16* __restrict__ Qb, const __bf16* __restrict__ Kb,
    const __bf16* __restrict__ Vt, __bf16* __restrict__ Ob) {
  __shared__ alignas(16) __bf16 Ps[8][16][40];  // per-wave P tile, 80B rows

  const int lane = threadIdx.x & 31;
  const int wv   = threadIdx.x >> 5;
  const int hf   = lane >> 4;
  const int l16  = lane & 15;
  const int wid  = blockIdx.x * 8 + wv;
  const int h    = wid & (NH - 1);
  const int qt   = wid >> 5;          // all 8 waves of a block share qt
  const int kvh  = h >> 2;
  const float scale = 0.08838834764831845f;  // 1/sqrt(128)

  // Q fragments: 4 d-chunks of 32
  bf16x16 aq[4];
  {
    const __bf16* qp = Qb + (size_t)(qt * 16 + l16) * H_DIM + h * HD + hf * 8;
#pragma unroll
    for (int c = 0; c < 4; ++c)
      aq[c] = concat8(*(const bf16x8*)(qp + c * 32), *(const bf16x8*)(qp + c * 32 + 16));
  }

  f32x8 o[8];
#pragma unroll
  for (int dt = 0; dt < 8; ++dt) o[dt] = zero8();
  float mrow[8], lrow[8];
#pragma unroll
  for (int r = 0; r < 8; ++r) { mrow[r] = -3e38f; lrow[r] = 0.0f; }

  const int nchunks = qt / 2 + 1;  // 32-key chunks touching the causal band
  for (int it = 0; it < nchunks; ++it) {
    const int base = it * 32;

    // two 16x16 score tiles
    f32x8 sc[2];
#pragma unroll
    for (int tile = 0; tile < 2; ++tile) {
      f32x8 s = zero8();
      const __bf16* kp = Kb + (size_t)(base + tile * 16 + l16) * (NKV * HD) + kvh * HD + hf * 8;
#pragma unroll
      for (int c = 0; c < 4; ++c) {
        bf16x16 bk = concat8(*(const bf16x8*)(kp + c * 32), *(const bf16x8*)(kp + c * 32 + 16));
        s = __builtin_amdgcn_wmma_f32_16x16x32_bf16(
            false, aq[c], false, bk, (short)0, s, false, false);
      }
      sc[tile] = s;
    }

    // scale + causal mask + row max (row m = hf*8+r spans the 16 lanes of hf)
    const int key0 = base + l16, key1 = base + 16 + l16;
    float rmax[8];
#pragma unroll
    for (int r = 0; r < 8; ++r) {
      int qg = qt * 16 + hf * 8 + r;
      float v0 = sc[0][r] * scale + ((key0 <= qg) ? 0.0f : NEG_INF);
      float v1 = sc[1][r] * scale + ((key1 <= qg) ? 0.0f : NEG_INF);
      sc[0][r] = v0; sc[1][r] = v1;
      rmax[r] = fmaxf(v0, v1);
    }
    for (int mask = 1; mask < 16; mask <<= 1)
#pragma unroll
      for (int r = 0; r < 8; ++r) rmax[r] = fmaxf(rmax[r], __shfl_xor(rmax[r], mask, 32));

    float sfac[8], psum[8];
#pragma unroll
    for (int r = 0; r < 8; ++r) {
      float mn = fmaxf(mrow[r], rmax[r]);
      sfac[r] = expf(mrow[r] - mn);
      mrow[r] = mn;
      float p0 = expf(sc[0][r] - mn);
      float p1 = expf(sc[1][r] - mn);
      psum[r] = p0 + p1;
      Ps[wv][hf * 8 + r][l16]      = (__bf16)p0;
      Ps[wv][hf * 8 + r][16 + l16] = (__bf16)p1;
    }
    for (int mask = 1; mask < 16; mask <<= 1)
#pragma unroll
      for (int r = 0; r < 8; ++r) psum[r] += __shfl_xor(psum[r], mask, 32);
#pragma unroll
    for (int r = 0; r < 8; ++r) lrow[r] = lrow[r] * sfac[r] + psum[r];
#pragma unroll
    for (int dt = 0; dt < 8; ++dt)
#pragma unroll
      for (int r = 0; r < 8; ++r) o[dt][r] *= sfac[r];

    // P A-fragment from LDS (layout matches A: elem e -> k = (e/8)*16+hf*8+e%8)
    const __bf16* pp = &Ps[wv][l16][hf * 8];
    bf16x16 ap = concat8(*(const bf16x8*)pp, *(const bf16x8*)(pp + 16));

    // O += P @ V   (Vt makes B-fragment reads contiguous: elem e -> key)
#pragma unroll
    for (int dt = 0; dt < 8; ++dt) {
      const __bf16* vp = Vt + ((size_t)kvh * HD + dt * 16 + l16) * S_LEN + base + hf * 8;
      bf16x16 bv = concat8(*(const bf16x8*)vp, *(const bf16x8*)(vp + 16));
      o[dt] = __builtin_amdgcn_wmma_f32_16x16x32_bf16(
          false, ap, false, bv, (short)0, o[dt], false, false);
    }
  }

  // normalize and store O as bf16 (S, NH*HD)
  float invl[8];
#pragma unroll
  for (int r = 0; r < 8; ++r) invl[r] = 1.0f / lrow[r];
#pragma unroll
  for (int dt = 0; dt < 8; ++dt)
#pragma unroll
    for (int r = 0; r < 8; ++r)
      Ob[(size_t)(qt * 16 + hf * 8 + r) * H_DIM + h * HD + dt * 16 + l16] =
          (__bf16)(o[dt][r] * invl[r]);
}

// ---------------------------------------------------------------------------
extern "C" void kernel_launch(void* const* d_in, const int* in_sizes, int n_in,
                              void* d_out, int out_size, void* d_ws, size_t ws_size,
                              hipStream_t stream) {
  (void)in_sizes; (void)n_in; (void)out_size; (void)ws_size;
  const float* hidden = (const float*)d_in[0];
  // d_in[1] attention_mask: causal -1e9 mask, applied analytically in-kernel
  const int*   pos = (const int*)d_in[2];
  const float* Wq  = (const float*)d_in[3];
  const float* Wk  = (const float*)d_in[4];
  const float* Wv  = (const float*)d_in[5];
  const float* Wo  = (const float*)d_in[6];
  float* out = (float*)d_out;

  constexpr size_t N_X  = (size_t)S_LEN * H_DIM;          // 8388608
  constexpr size_t N_WQ = (size_t)H_DIM * (NH * HD);      // 16777216
  constexpr size_t N_WK = (size_t)H_DIM * (NKV * HD);     // 4194304
  constexpr size_t N_KV = (size_t)S_LEN * (NKV * HD);     // 2097152

  __bf16* p    = (__bf16*)d_ws;
  __bf16* Xbf  = p; p += N_X;
  __bf16* Wqb  = p; p += N_WQ;
  __bf16* Wkb  = p; p += N_WK;
  __bf16* Wvb  = p; p += N_WK;
  __bf16* Wob  = p; p += N_WQ;
  __bf16* Qb   = p; p += N_X;
  __bf16* Kb   = p; p += N_KV;
  __bf16* Vraw = p; p += N_KV;
  __bf16* Vt   = p; p += N_KV;
  __bf16* Ob   = p; p += N_X;

  auto cvt = [&](const float* src, __bf16* dst, size_t n) {
    int n4 = (int)(n / 4);
    cvt_f32_bf16<<<(n4 + 255) / 256, 256, 0, stream>>>(src, dst, n4);
  };
  cvt(hidden, Xbf, N_X);
  cvt(Wq, Wqb, N_WQ);
  cvt(Wk, Wkb, N_WK);
  cvt(Wv, Wvb, N_WK);
  cvt(Wo, Wob, N_WQ);

  // QKV projections (bf16 out)
  gemm_bf16_kernel<false><<<dim3(S_LEN / 128, H_DIM / 64), 256, 0, stream>>>(
      Xbf, Wqb, nullptr, Qb, S_LEN, H_DIM, H_DIM);
  gemm_bf16_kernel<false><<<dim3(S_LEN / 128, (NKV * HD) / 64), 256, 0, stream>>>(
      Xbf, Wkb, nullptr, Kb, S_LEN, NKV * HD, H_DIM);
  gemm_bf16_kernel<false><<<dim3(S_LEN / 128, (NKV * HD) / 64), 256, 0, stream>>>(
      Xbf, Wvb, nullptr, Vraw, S_LEN, NKV * HD, H_DIM);

  // RoPE (in place) + V transpose
  rope_kernel<<<(S_LEN * NH * 64) / 256, 256, 0, stream>>>(Qb, pos, NH);
  rope_kernel<<<(S_LEN * NKV * 64) / 256, 256, 0, stream>>>(Kb, pos, NKV);
  v_transpose_kernel<<<(S_LEN * NKV * HD) / 256, 256, 0, stream>>>(Vraw, Vt);

  // attention: 128 q-tiles * 32 heads = 4096 waves, 8 waves/block
  flash_attn_kernel<<<(128 * NH) / 8, 256, 0, stream>>>(Qb, Kb, Vt, Ob);

  // output projection (f32 out)
  gemm_bf16_kernel<true><<<dim3(S_LEN / 128, H_DIM / 64), 256, 0, stream>>>(
      Ob, Wob, out, nullptr, S_LEN, H_DIM, H_DIM);
}
